// VanillaRNNModel_10711648436258
// MI455X (gfx1250) — matrix operand order repaired and checked by reading, not compiled, not run on hardware
//
#include <hip/hip_runtime.h>
#include <hip/hip_bf16.h>

// Elman RNN scan for MI455X (gfx1250, wave32).
// B=4096, T=2048, I=8, H=10, O=1.
//
// One wave owns a 16-batch tile and runs the whole T=2048 recurrence with
// one V_WMMA_F32_16X16X32_F16 per step:
//   A (loop-invariant) = [W_ih | W_hh] padded to 16x32 f16
//   B (per step)       = [x_t | h_{t-1}] padded to 32x16 f16 (columns = batches)
//   C (loop-invariant) = (b_ih + b_hh) broadcast
//   h_t = tanh(D)
// Memory floor: 268 MB of x / 23.3 TB/s ~= 11.5 us; everything else is noise.
//
// Codegen notes (from round-2 asm): use ext_vector float4 so prefetches stay
// single global_load_b128; immediate-offset loads + one pointer bump per
// 8-step block; last block peeled so no clamp / per-step address math.

#define B_SZ 4096
#define T_SZ 2048
#define I_SZ 8
#define H_SZ 10

typedef __attribute__((ext_vector_type(16))) _Float16 v16h;
typedef __attribute__((ext_vector_type(8)))  float    v8f;
typedef __attribute__((ext_vector_type(4)))  float    v4f;

union BS { v16h v; unsigned u[8]; };

__device__ __forceinline__ unsigned pk_f16(float a, float b) {
#if __has_builtin(__builtin_amdgcn_cvt_pkrtz)
    // returns __fp16 ext_vector(2); bit-cast straight to u32 (one VGPR)
    return __builtin_bit_cast(unsigned, __builtin_amdgcn_cvt_pkrtz(a, b));
#else
    union { _Float16 h[2]; unsigned u; } v;
    v.h[0] = (_Float16)a; v.h[1] = (_Float16)b;
    return v.u;
#endif
}

__device__ __forceinline__ float tanh_fast(float x) {
#if __has_builtin(__builtin_amdgcn_tanhf)
    return __builtin_amdgcn_tanhf(x);           // gfx1250 v_tanh_f32 (trans, co-exec)
#else
    float e = __expf(2.0f * x);
#if __has_builtin(__builtin_amdgcn_rcpf)
    return 1.0f - 2.0f * __builtin_amdgcn_rcpf(e + 1.0f);
#else
    return 1.0f - 2.0f / (e + 1.0f);
#endif
#endif
}

// Rebuild the B operand (state matrix) for the next step.
// Lane layout (B-matrix 32x16 f16, contiguous k order, same as aW):
//   lanes 0-15  (col b): v0..v3 = K0..7  = x_t[0..7], v4..v7 = K8..15 = h[0..7]
//   lanes 16-31 (col b): v0     = K16,17 = h[8],h[9],  v1..v7 = 0
// D landed as: lane b v0..7 = h[0..7]; lane b+16 v0,v1 = h[8],h[9]  -> same-lane packs.
// x-row split: lane b loaded x[0..3], lane b+16 loaded x[4..7] -> 2 half-swap shuffles.
__device__ __forceinline__ void build_state(BS& s, v4f xf,
                                            const float (&th)[8], bool hi) {
    unsigned q0  = pk_f16(xf.x, xf.y);
    unsigned q1  = pk_f16(xf.z, xf.w);
    unsigned xq0 = __shfl_xor(q0, 16, 32);      // partner's x-half
    unsigned xq1 = __shfl_xor(q1, 16, 32);
    unsigned p0  = pk_f16(th[0], th[1]);
    unsigned p1  = pk_f16(th[2], th[3]);
    unsigned p2  = pk_f16(th[4], th[5]);
    unsigned p3  = pk_f16(th[6], th[7]);
    s.u[0] = hi ? p0 : q0;   // hi: pack(h8,h9) (K16,17) | lo: x0,x1 (K0,1)
    s.u[1] = hi ? 0u : q1;   // lo: x2,x3
    s.u[2] = hi ? 0u : xq0;  // lo: x4,x5 (from partner)
    s.u[3] = hi ? 0u : xq1;  // lo: x6,x7 (from partner)
    s.u[4] = hi ? 0u : p0;   // lo: h0,h1 (K8,9)
    s.u[5] = hi ? 0u : p1;
    s.u[6] = hi ? 0u : p2;
    s.u[7] = hi ? 0u : p3;
}

__global__ __launch_bounds__(256) void rnn_scan_wmma(
    const float* __restrict__ x,    const float* __restrict__ W_ih,
    const float* __restrict__ b_ih, const float* __restrict__ W_hh,
    const float* __restrict__ b_hh, const float* __restrict__ W_hy,
    const float* __restrict__ b_hy, float* __restrict__ out)
{
    const int  lane  = threadIdx.x & 31;
    const int  wave  = (blockIdx.x * blockDim.x + threadIdx.x) >> 5;
    const int  m     = lane & 15;
    const bool hi    = lane >= 16;
    const int  batch = wave * 16 + m;

    // ---- A operand: 16x32 f16, row m = [W_ih[m][0..7], W_hh[m][0..9], 0...] ----
    // A layout (16-bit 16x32, contiguous k): lanes 0-15 e0..15 -> K0..15;
    //                          lanes 16-31 e0..15 -> K16..31.
    v16h aW;
    #pragma unroll
    for (int i = 0; i < 16; ++i) aW[i] = (_Float16)0.0f;
    if (m < H_SZ) {
        if (!hi) {
            #pragma unroll
            for (int k = 0; k < 8; ++k) aW[k] = (_Float16)W_ih[m * I_SZ + k]; // K=0..7
            #pragma unroll
            for (int k = 0; k < 8; ++k) aW[8 + k] = (_Float16)W_hh[m * H_SZ + k]; // K=8..15 -> h0..7
        } else {
            aW[0] = (_Float16)W_hh[m * H_SZ + 8];                            // K=16 -> h8
            aW[1] = (_Float16)W_hh[m * H_SZ + 9];                            // K=17 -> h9
        }
    }

    // ---- C operand: bias broadcast (C[j][b] = b_ih[j]+b_hh[j]) ----
    v8f cB;
    #pragma unroll
    for (int j = 0; j < 8; ++j) cB[j] = 0.0f;
    if (!hi) {
        #pragma unroll
        for (int j = 0; j < 8; ++j) cB[j] = b_ih[j] + b_hh[j];
    } else {
        cB[0] = b_ih[8] + b_hh[8];
        cB[1] = b_ih[9] + b_hh[9];
    }

    // ---- x stream: lane b -> x[batch][t][0..3], lane b+16 -> x[batch][t][4..7]
    // One row = 32 bytes; per-lane running base bumped once per 8-step block,
    // every load uses a compile-time immediate offset.
    const char* xcur = reinterpret_cast<const char*>(x)
                     + ((size_t)batch * (T_SZ * 32)) + (hi ? 16 : 0);

    // 8-deep register prefetch ring: ~8 steps of compute covers load latency.
    v4f buf[8];
    #pragma unroll
    for (int j = 0; j < 8; ++j)
        buf[j] = *reinterpret_cast<const v4f*>(xcur + j * 32);

    float th[8];
    #pragma unroll
    for (int j = 0; j < 8; ++j) th[j] = 0.0f;   // h_{-1} = 0

    BS s;
    build_state(s, buf[0], th, hi);             // B_0 = [x_0 | 0]

    // ---- main loop: blocks of 8 steps; last block peeled (no prefetch) ----
    for (int tb = 0; tb + 8 < T_SZ; tb += 8) {
        #pragma unroll
        for (int k = 0; k < 8; ++k) {
            v4f xn = buf[(k + 1) & 7];           // x_{t+1} (k=7: written at k=0 this block)
            buf[k] = *reinterpret_cast<const v4f*>(xcur + (k + 8) * 32); // x_{t+8}

            v8f d = __builtin_amdgcn_wmma_f32_16x16x32_f16(
                false, aW, false, s.v, (short)0, cB, false, false);

            #pragma unroll
            for (int j = 0; j < 8; ++j) th[j] = tanh_fast(d[j]);

            build_state(s, xn, th, hi);          // B_{t+1}
        }
        xcur += 8 * 32;                          // one pointer bump per block
    }

    // ---- epilogue block: t = T-8 .. T-1, consume ring only ----
    #pragma unroll
    for (int k = 0; k < 8; ++k) {
        v8f d = __builtin_amdgcn_wmma_f32_16x16x32_f16(
            false, aW, false, s.v, (short)0, cB, false, false);
        #pragma unroll
        for (int j = 0; j < 8; ++j) th[j] = tanh_fast(d[j]);
        if (k < 7) build_state(s, buf[k + 1], th, hi);   // no state needed after t=T-1
    }

    // ---- y[b] = W_hy . h_last[b] + b_hy ----
    // lanes 0-15 hold h[0..7], lanes 16-31 hold h[8],h[9] (th[2..7]=tanh(0)=0 there).
    float why[8];
    #pragma unroll
    for (int j = 0; j < 8; ++j) {
        int idx = (hi ? 8 : 0) + j;
        why[j] = (idx < H_SZ) ? W_hy[idx] : 0.0f;
    }
    float acc = 0.0f;
    #pragma unroll
    for (int j = 0; j < 8; ++j) acc += why[j] * th[j];
    acc += __shfl_xor(acc, 16, 32);
    if (!hi) out[batch] = acc + b_hy[0];
}

extern "C" void kernel_launch(void* const* d_in, const int* in_sizes, int n_in,
                              void* d_out, int out_size, void* d_ws, size_t ws_size,
                              hipStream_t stream) {
    const float* x    = (const float*)d_in[0];
    const float* W_ih = (const float*)d_in[1];
    const float* b_ih = (const float*)d_in[2];
    const float* W_hh = (const float*)d_in[3];
    const float* b_hh = (const float*)d_in[4];
    const float* W_hy = (const float*)d_in[5];
    const float* b_hy = (const float*)d_in[6];
    float* out = (float*)d_out;

    // 256 chains (16 batches each), 1 wave per chain, 8 waves per block.
    dim3 block(256);
    dim3 grid(B_SZ / 16 / 8);   // 32 blocks * 8 waves = 256 waves
    rnn_scan_wmma<<<grid, block, 0, stream>>>(x, W_ih, b_ih, W_hh, b_hh, W_hy, b_hy, out);
}
